// ELocalLayer_18786186953537
// MI455X (gfx1250) — matrix operand order repaired and checked by reading, not compiled, not run on hardware
//
#include <hip/hip_runtime.h>
#include <hip/hip_bf16.h>

// Problem constants (fixed by the reference):
//   T = 256 samples, B = 16, N_U = 16, n = 12 orbitals, f = 14 features.
// h2 (21 MB) dominates traffic -> streaming kernel k2 is the roofline piece.
// k1 does per-sample algebra incl. S1 = s1f * s1f^T via v_wmma_f32_16x16x32_bf16
// (exact: s1f entries are small integers, bf16-representable).

#define NT   256     // total samples
#define NB   16      // batch
#define NU   16      // walkers per batch
#define NO   12      // orbitals
#define NF   14      // features
#define NC1  144     // n^2 one-body columns
#define NC2  20736   // n^4 two-body columns
#define KPAD 160     // 144 padded to multiple of 32 for bf16 WMMA

typedef __attribute__((ext_vector_type(16))) __bf16 v16bf;
typedef __attribute__((ext_vector_type(8)))  float  v8f;

// float -> bf16 bits, round-to-nearest-even (exact for our small integers)
__device__ inline unsigned short f2bf_bits(float f) {
    union { float f; unsigned u; } v; v.f = f;
    unsigned r = v.u + 0x7FFFu + ((v.u >> 16) & 1u);
    return (unsigned short)(r >> 16);
}

// 12x12 determinant, Gaussian elimination w/ partial pivoting, in LDS.
__device__ float det12(float* M) {
    float det = 1.0f;
    for (int c = 0; c < NO; ++c) {
        int piv = c; float best = fabsf(M[c * NO + c]);
        for (int r = c + 1; r < NO; ++r) {
            float v = fabsf(M[r * NO + c]);
            if (v > best) { best = v; piv = r; }
        }
        if (piv != c) {
            for (int j = c; j < NO; ++j) {
                float tmp = M[c * NO + j]; M[c * NO + j] = M[piv * NO + j]; M[piv * NO + j] = tmp;
            }
            det = -det;
        }
        float p = M[c * NO + c];
        det *= p;
        if (p != 0.0f) {
            float inv = 1.0f / p;
            for (int r = c + 1; r < NO; ++r) {
                float fac = M[r * NO + c] * inv;
                for (int j = c + 1; j < NO; ++j) M[r * NO + j] -= fac * M[c * NO + j];
            }
        }
    }
    return det;
}

// ---------------------------------------------------------------------------
// Kernel 1: one wave (32 lanes) per sample. Computes G, s1f, S1 (WMMA),
// det-based e1 and p, and per-sample u / udotx / x for the streaming kernel.
// ---------------------------------------------------------------------------
__global__ __launch_bounds__(32) void k1_per_sample(
    const float* __restrict__ mfv,   // (T, 12, 14)
    const float* __restrict__ h1,    // (T, 144)
    const int*   __restrict__ vnhf,  // (T, 12)
    const float* __restrict__ w,     // (14, 14)
    float* ws_e1, float* ws_p, float* ws_u, float* ws_ud, int* ws_x)
{
    __shared__ float sW[16 * 16];          // Wsym, padded
    __shared__ float sPhi[16 * 16];        // phi, padded
    __shared__ float sH[16 * 16];          // H = Wsym @ phi^T : [f][orb]
    __shared__ float sG[16 * 16];          // G
    __shared__ unsigned short sA[16 * KPAD]; // s1f as bf16 bits: [orb m][col k]
    __shared__ float sS1[16 * 16];         // s1f @ s1f^T
    __shared__ float sM1[NO * NO];
    __shared__ float sM2[NO * NO];
    __shared__ float sU[NO];
    __shared__ int   sX[NO];

    const int t    = blockIdx.x;
    const int lane = threadIdx.x;

    for (int i = lane; i < 256; i += 32) { sW[i] = 0.f; sPhi[i] = 0.f; sH[i] = 0.f; }
    for (int i = lane; i < 16 * KPAD; i += 32) sA[i] = 0;
    __syncthreads();

    // Wsym = w + w^T  (14x14)
    for (int i = lane; i < NF * NF; i += 32) {
        int r = i / NF, c = i % NF;
        sW[r * 16 + c] = w[r * NF + c] + w[c * NF + r];
    }
    // phi (12x14)
    const float* phi = mfv + (size_t)t * NO * NF;
    for (int i = lane; i < NO * NF; i += 32) {
        int r = i / NF, c = i % NF;
        sPhi[r * 16 + c] = phi[i];
    }
    if (lane < NO) sX[lane] = vnhf[t * NO + lane];
    __syncthreads();

    // u[k] = sum_f phi[k,f]
    if (lane < NO) {
        float s = 0.f;
        for (int f = 0; f < NF; ++f) s += sPhi[lane * 16 + f];
        sU[lane] = s;
    }
    // H[f][o] = sum_g Wsym[f][g] * phi[o][g]   (14x12)
    for (int i = lane; i < NF * NO; i += 32) {
        int f = i / NO, o = i % NO;
        float s = 0.f;
        for (int g = 0; g < NF; ++g) s += sW[f * 16 + g] * sPhi[o * 16 + g];
        sH[f * 16 + o] = s;
    }
    __syncthreads();
    // G[a][o] = sum_f phi[a][f] * H[f][o]   (12x12, exact f32)
    for (int i = lane; i < NO * NO; i += 32) {
        int a = i / NO, o = i % NO;
        float s = 0.f;
        for (int f = 0; f < NF; ++f) s += sPhi[a * 16 + f] * sH[f * 16 + o];
        sG[a * 16 + o] = s;
    }

    // ---- one-body columns: mask, s1f tile (bf16), Sum h1^2 over mask ----
    const float* h1t = h1 + (size_t)t * NC1;
    float h1sum = 0.f;
    for (int col = lane; col < NC1; col += 32) {
        int a = col / NO, o = col % NO;
        int sv[NO]; int mn = 100, mx = -100;
        #pragma unroll
        for (int k = 0; k < NO; ++k) {
            int s = sX[k] + (int)(k == o) - (int)(k == a);
            sv[k] = s; mn = min(mn, s); mx = max(mx, s);
        }
        float hv = h1t[col];
        if (((mx - mn) == 1) && (hv > 0.f)) {
            h1sum += hv * hv;
            #pragma unroll
            for (int k = 0; k < NO; ++k) sA[k * KPAD + col] = f2bf_bits((float)sv[k]);
        }
    }
    for (int off = 16; off > 0; off >>= 1) h1sum += __shfl_xor(h1sum, off);
    __syncthreads();

    // ---- S1 = s1f @ s1f^T via 5x v_wmma_f32_16x16x32_bf16 (K padded to 160) ----
    {
        v8f acc = {};
        const int m   = lane & 15;
        const int hiA = (lane >= 16) ? 8  : 0;  // A-layout lane-half K offset
        const int hiB = (lane >= 16) ? 16 : 0;  // unused; B loads use hiA too
        for (int c = 0; c < KPAD / 32; ++c) {
            const int kb = 32 * c;
            unsigned short ah[16], bh[16];
            #pragma unroll
            for (int h = 0; h < 16; ++h) {
                int ka = kb + h + ((h >= 8) ? 8 : 0) + hiA; // A: 16-bit 16xK layout
                int kbb = kb + h + ((h >= 8) ? 8 : 0) + hiA; // B: same k order as A
                ah[h] = sA[m * KPAD + ka];   // A[m][k] = s1f[m][k]
                bh[h] = sA[m * KPAD + kbb];  // B[k][n] = s1f[n][k], n == lane&15
            }
            v16bf afr, bfr;
            __builtin_memcpy(&afr, ah, sizeof(afr));
            __builtin_memcpy(&bfr, bh, sizeof(bfr));
            acc = __builtin_amdgcn_wmma_f32_16x16x32_bf16(
                false, afr, false, bfr, (short)0, acc, false, false);
        }
        #pragma unroll
        for (int r = 0; r < 8; ++r) {
            int M = r + ((lane >= 16) ? 8 : 0);      // C/D layout
            sS1[M * 16 + (lane & 15)] = acc[r];
        }
    }
    __syncthreads();

    // ---- Hadamard products + determinants ----
    for (int i = lane; i < NO * NO; i += 32) {
        int a = i / NO, o = i % NO;
        float g = sG[a * 16 + o];
        sM1[i] = g * sS1[a * 16 + o];
        sM2[i] = g * (float)(sX[a] * sX[o]);
    }
    __syncthreads();
    if (lane == 0) {
        float d1 = det12(sM1);
        float d2 = det12(sM2);
        ws_e1[t] = d1 * d1 * h1sum;
        ws_p[t]  = d2 * d2;
        float ud = 0.f;
        for (int k = 0; k < NO; ++k) ud += sU[k] * (float)sX[k];
        ws_ud[t] = ud;
    }
    if (lane < NO) { ws_u[t * NO + lane] = sU[lane]; ws_x[t * NO + lane] = sX[lane]; }
}

// ---------------------------------------------------------------------------
// Kernel 2: the roofline piece — stream 21 MB of h2 once (coalesced,
// prefetched), masked accumulate of (h2 * a2)^2 where a2 is closed-form.
// One 256-thread block per sample; 81 columns per thread.
// ---------------------------------------------------------------------------
__global__ __launch_bounds__(256) void k2_two_body(
    const float* __restrict__ h2,
    const float* __restrict__ ws_e1, const float* __restrict__ ws_u,
    const float* __restrict__ ws_ud, const int* __restrict__ ws_x,
    float* ws_e)
{
    __shared__ float su[NO];
    __shared__ int   sx[NO];
    __shared__ float sud;
    __shared__ float red[256];

    const int t   = blockIdx.x;
    const int tid = threadIdx.x;
    if (tid < NO) { su[tid] = ws_u[t * NO + tid]; sx[tid] = ws_x[t * NO + tid]; }
    if (tid == 0) sud = ws_ud[t];
    __syncthreads();

    int xk[NO];
    #pragma unroll
    for (int k = 0; k < NO; ++k) xk[k] = sx[k];
    const float ud = sud;

    const float* h2t = h2 + (size_t)t * NC2;
    float acc = 0.f;
    for (int col = tid; col < NC2; col += 256) {
        if (col + 2048 < NC2) __builtin_prefetch(h2t + col + 2048, 0, 1);
        float hv = h2t[col];
        if (hv > 0.f) {
            int P = col / NC1, Q = col - P * NC1;       // col = P*144 + Q
            int aP = P / NO, iP = P - aP * NO;
            int aQ = Q / NO, iQ = Q - aQ * NO;
            int mn = 100, mx = -100;
            #pragma unroll
            for (int k = 0; k < NO; ++k) {
                int s = xk[k] + (int)(k == iQ) - (int)(k == aQ)
                              - (int)(k == iP) + (int)(k == aP);
                mn = min(mn, s); mx = max(mx, s);
            }
            if ((mx - mn) == 1) {
                float a2 = ud + su[iQ] - su[aQ] - su[iP] + su[aP];
                float v = hv * a2;
                acc += v * v;
            }
        }
    }
    red[tid] = acc;
    __syncthreads();
    for (int s = 128; s > 0; s >>= 1) {
        if (tid < s) red[tid] += red[tid + s];
        __syncthreads();
    }
    if (tid == 0) ws_e[t] = ws_e1[t] + red[0];
}

// ---------------------------------------------------------------------------
// Kernel 3: per-batch weighted average: out[b] = sum_u e*p / sum_u p
// ---------------------------------------------------------------------------
__global__ __launch_bounds__(256) void k3_combine(
    const float* __restrict__ ws_e, const float* __restrict__ ws_p,
    float* __restrict__ out)
{
    __shared__ float sep[NT];
    __shared__ float spp[NT];
    const int tid = threadIdx.x;
    float e = ws_e[tid], p = ws_p[tid];
    sep[tid] = e * p;
    spp[tid] = p;
    __syncthreads();
    if (tid < NB) {
        float num = 0.f, den = 0.f;
        for (int u = 0; u < NU; ++u) {
            num += sep[tid * NU + u];
            den += spp[tid * NU + u];
        }
        out[tid] = num / den;
    }
}

extern "C" void kernel_launch(void* const* d_in, const int* in_sizes, int n_in,
                              void* d_out, int out_size, void* d_ws, size_t ws_size,
                              hipStream_t stream) {
    const float* mfv = (const float*)d_in[0];   // (T,12,14) f32
    const float* h1  = (const float*)d_in[1];   // (T,144)   f32
    const float* h2  = (const float*)d_in[2];   // (T,20736) f32
    const int*   vn  = (const int*)d_in[3];     // (B,NU,12) i32
    const float* w   = (const float*)d_in[4];   // (14,14)   f32

    // Workspace layout (floats / ints), ~28 KB total.
    float* ws_e1 = (float*)d_ws;
    float* ws_p  = ws_e1 + NT;
    float* ws_u  = ws_p  + NT;
    float* ws_ud = ws_u  + NT * NO;
    int*   ws_x  = (int*)(ws_ud + NT);
    float* ws_e  = (float*)(ws_x + NT * NO);

    k1_per_sample<<<NT, 32, 0, stream>>>(mfv, h1, vn, w, ws_e1, ws_p, ws_u, ws_ud, ws_x);
    k2_two_body  <<<NT, 256, 0, stream>>>(h2, ws_e1, ws_u, ws_ud, ws_x, ws_e);
    k3_combine   <<<1, NT, 0, stream>>>(ws_e, ws_p, (float*)d_out);
}
